// ArcMarginSoftMaxLoss_79568564126306
// MI455X (gfx1250) — compile-verified, matrix operand order checked by reading, not run on hardware
//
#include <hip/hip_runtime.h>
#include <hip/hip_bf16.h>
#include <math.h>

#define EMB   192
#define NCLS  5994
#define NROW  8192
#define CPAD  6000      /* 375 * 16 */
#define NTILE 375

__device__ __constant__ float kCOS_M = 0.9800665778412416f;   /* cos(0.2)        */
__device__ __constant__ float kSIN_M = 0.19866933079506122f;  /* sin(0.2)        */
__device__ __constant__ float kTH    = -0.9800665778412416f;  /* cos(pi-0.2)     */
__device__ __constant__ float kMM    = 0.03973386615901224f;  /* sin(pi-0.2)*0.2 */
#define S_SCALE 30.0f

typedef __attribute__((ext_vector_type(16))) __bf16 v16bf;
typedef __attribute__((ext_vector_type(8)))  __bf16 v8bf;
typedef __attribute__((ext_vector_type(8)))  float  v8f;

static __device__ __forceinline__ v16bf load_frag(const __bf16* __restrict__ p) {
    // Two contiguous 16-byte chunks per lane: K = base..base+7 and K = base+16..base+23
    v8bf a = *(const v8bf*)(p);
    v8bf b = *(const v8bf*)(p + 16);
    return __builtin_shufflevector(a, b, 0,1,2,3,4,5,6,7,8,9,10,11,12,13,14,15);
}

// ---------------------------------------------------------------------------
// Kernel 1: L2-normalize rows, split each fp32 value into bf16 hi + bf16 lo.
// One wave32 per row (192 = 32 lanes x 6 elements). Rows >= valid_rows -> 0.
// ---------------------------------------------------------------------------
__global__ void norm_split_kernel(const float* __restrict__ src,
                                  __bf16* __restrict__ dhi,
                                  __bf16* __restrict__ dlo,
                                  int rows, int valid_rows) {
    int wave = (int)((blockIdx.x * blockDim.x + threadIdx.x) >> 5);
    int lane = threadIdx.x & 31;
    if (wave >= rows) return;

    float v[6];
    float ss = 0.0f;
    if (wave < valid_rows) {
        const float* p = src + (size_t)wave * EMB;
#pragma unroll
        for (int j = 0; j < 6; ++j) {
            v[j] = p[lane + 32 * j];
            ss += v[j] * v[j];
        }
    } else {
#pragma unroll
        for (int j = 0; j < 6; ++j) v[j] = 0.0f;
    }
#pragma unroll
    for (int m = 16; m >= 1; m >>= 1) ss += __shfl_xor(ss, m, 32);

    float nrm = fmaxf(sqrtf(ss), 1e-12f);

    __bf16* ph = dhi + (size_t)wave * EMB;
    __bf16* pl = dlo + (size_t)wave * EMB;
#pragma unroll
    for (int j = 0; j < 6; ++j) {
        float r  = v[j] / nrm;           // matches reference v / max(||v||, eps)
        __bf16 h = (__bf16)r;            // RNE
        float rem = r - (float)h;
        __bf16 l = (__bf16)rem;
        ph[lane + 32 * j] = h;
        pl[lane + 32 * j] = l;
    }
}

// ---------------------------------------------------------------------------
// Kernel 2: fused cosine GEMM (split bf16, 3x WMMA) + online arc-softmax.
// Block = 128 threads = 4 waves; each wave owns a 16-row M-tile and streams
// all 375 column tiles of W. W (hi+lo ~4.6MB) is L2-resident on MI455X.
// ---------------------------------------------------------------------------
__launch_bounds__(128, 1)
__global__ void arc_fused_kernel(const __bf16* __restrict__ xh,
                                 const __bf16* __restrict__ xl,
                                 const __bf16* __restrict__ wh,
                                 const __bf16* __restrict__ wl,
                                 const int*    __restrict__ labels,
                                 float* __restrict__ loss_out,
                                 float* __restrict__ corr_out) {
    const int lane  = threadIdx.x & 31;
    const int wid   = threadIdx.x >> 5;                 // 0..3
    const int mbase = ((int)blockIdx.x * 4 + wid) * 16; // 16 rows per wave
    const int h     = lane >> 4;                        // lane half (K-range select)
    const int m16   = lane & 15;

    // --- A fragments: row = mbase + m16, all 6 K-slabs, hi and lo ---
    const __bf16* pxh = xh + (size_t)(mbase + m16) * EMB + 8 * h;
    const __bf16* pxl = xl + (size_t)(mbase + m16) * EMB + 8 * h;
    v16bf ahf[6], alf[6];
#pragma unroll
    for (int s = 0; s < 6; ++s) {
        ahf[s] = load_frag(pxh + 32 * s);
        alf[s] = load_frag(pxl + 32 * s);
    }

    // --- per-lane online state; VGPR r of D maps to row mbase + 8*h + r ---
    float ssum[8], tval[8], bmax[8];
    int   bidx[8], lab[8];
#pragma unroll
    for (int r = 0; r < 8; ++r) {
        ssum[r] = 0.0f;
        tval[r] = 0.0f;
        bmax[r] = -INFINITY;
        bidx[r] = 0x7fffffff;
        lab[r]  = labels[mbase + 8 * h + r];
    }

    for (int nt = 0; nt < NTILE; ++nt) {
        const int nbase = nt * 16;
        const int col   = nbase + m16;  // D layout: lane holds column nbase+(lane&15)
        const __bf16* pwh = wh + (size_t)col * EMB + 8 * h;
        const __bf16* pwl = wl + (size_t)col * EMB + 8 * h;

        v8f acc = {};
#pragma unroll
        for (int s = 0; s < 6; ++s) {
            v16bf bhf = load_frag(pwh + 32 * s);
            v16bf blf = load_frag(pwl + 32 * s);
            // split-precision product: hi*hi + hi*lo + lo*hi (lo*lo dropped, ~2^-16)
            acc = __builtin_amdgcn_wmma_f32_16x16x32_bf16(false, ahf[s], false, bhf,
                                                          (short)0, acc, false, false);
            acc = __builtin_amdgcn_wmma_f32_16x16x32_bf16(false, ahf[s], false, blf,
                                                          (short)0, acc, false, false);
            acc = __builtin_amdgcn_wmma_f32_16x16x32_bf16(false, alf[s], false, bhf,
                                                          (short)0, acc, false, false);
        }

        const bool valid = (col < NCLS);
#pragma unroll
        for (int r = 0; r < 8; ++r) {
            float v = acc[r];
            if (valid) {
                // cos in [-1,1] => S*(v-1) in [-60,0]: safe without online max
                ssum[r] += __expf(fmaf(S_SCALE, v, -S_SCALE));
                if (v > bmax[r]) { bmax[r] = v; bidx[r] = col; }
                if (col == lab[r]) tval[r] = v;
            }
        }
    }

    // --- reduce across the 16 lanes of each half (deterministic xor tree) ---
#pragma unroll
    for (int r = 0; r < 8; ++r) {
#pragma unroll
        for (int mask = 1; mask <= 8; mask <<= 1) {
            ssum[r] += __shfl_xor(ssum[r], mask, 32);
            tval[r] += __shfl_xor(tval[r], mask, 32);
            float ob = __shfl_xor(bmax[r], mask, 32);
            int   oi = __shfl_xor(bidx[r], mask, 32);
            if (ob > bmax[r] || (ob == bmax[r] && oi < bidx[r])) {
                bmax[r] = ob; bidx[r] = oi;
            }
        }
    }

    if (m16 == 0) {
#pragma unroll
        for (int r = 0; r < 8; ++r) {
            const int row = mbase + 8 * h + r;
            float ct   = tval[r];
            float sine = sqrtf(fmaxf(0.0f, 1.0f - ct * ct));
            float phi  = ct * kCOS_M - sine * kSIN_M;
            if (!(ct > kTH)) phi = ct - kMM;
            // swap exp(S*cos_t) -> exp(S*phi) inside the (shifted) partition sum
            float sp = ssum[r] - __expf(fmaf(S_SCALE, ct,  -S_SCALE))
                               + __expf(fmaf(S_SCALE, phi, -S_SCALE));
            float loss = (S_SCALE + logf(sp)) - S_SCALE * phi;
            loss_out[row] = loss;
            corr_out[row] = (bidx[r] == lab[r]) ? 1.0f : 0.0f;
        }
    }
}

// ---------------------------------------------------------------------------
// Kernel 3: deterministic final reduction -> d_out[0]=mean loss, d_out[1]=acc
// ---------------------------------------------------------------------------
__global__ void final_reduce_kernel(const float* __restrict__ loss,
                                    const float* __restrict__ corr,
                                    float* __restrict__ out) {
    __shared__ float sl[256];
    __shared__ float sc[256];
    const int tid = threadIdx.x;
    float a = 0.0f, b = 0.0f;
    for (int i = tid; i < NROW; i += 256) { a += loss[i]; b += corr[i]; }
    sl[tid] = a; sc[tid] = b;
    __syncthreads();
#pragma unroll
    for (int s = 128; s >= 1; s >>= 1) {
        if (tid < s) { sl[tid] += sl[tid + s]; sc[tid] += sc[tid + s]; }
        __syncthreads();
    }
    if (tid == 0) {
        out[0] = sl[0] / (float)NROW;
        out[1] = sc[0];
    }
}

extern "C" void kernel_launch(void* const* d_in, const int* in_sizes, int n_in,
                              void* d_out, int out_size, void* d_ws, size_t ws_size,
                              hipStream_t stream) {
    const float* x      = (const float*)d_in[0];
    const int*   labels = (const int*)  d_in[1];
    const float* W      = (const float*)d_in[2];
    float* out = (float*)d_out;
    (void)in_sizes; (void)n_in; (void)out_size; (void)ws_size;

    // workspace layout (bf16 counts): xh, xl [NROW*EMB] ; wh, wl [CPAD*EMB] ; then floats
    __bf16* xh = (__bf16*)d_ws;
    __bf16* xl = xh + (size_t)NROW * EMB;
    __bf16* wh = xl + (size_t)NROW * EMB;
    __bf16* wl = wh + (size_t)CPAD * EMB;
    float*  loss_buf = (float*)(wl + (size_t)CPAD * EMB);
    float*  corr_buf = loss_buf + NROW;

    // normalize + split x (8192 rows, 8 waves/block)
    norm_split_kernel<<<(NROW * 32 + 255) / 256, 256, 0, stream>>>(x, xh, xl, NROW, NROW);
    // normalize + split W, pad to 6000 rows with zeros
    norm_split_kernel<<<(CPAD * 32 + 255) / 256, 256, 0, stream>>>(W, wh, wl, CPAD, NCLS);

    // fused cosine GEMM + arc-margin online softmax: 128 blocks x 4 waves x 16 rows
    arc_fused_kernel<<<NROW / 64, 128, 0, stream>>>(xh, xl, wh, wl, labels,
                                                    loss_buf, corr_buf);

    final_reduce_kernel<<<1, 256, 0, stream>>>(loss_buf, corr_buf, out);
}
